// CorrespondenceGenerationArch_67430986547239
// MI455X (gfx1250) — matrix operand order repaired and checked by reading, not compiled, not run on hardware
//
#include <hip/hip_runtime.h>
#include <hip/hip_bf16.h>

// ---------------------------------------------------------------------------
// CorrespondenceGenerationArch for MI455X (gfx1250, wave32, WMMA)
// Stage 1: channel-normalize dense features -> f16, location-major (y,x,c)
// Stage 2: patch norms (sum of per-location sq-norms over 3x3)
// Stage 3: fused corr GEMM (6084x6084x2304, f16 WMMA, f32 accum) + argmax
//          - async global->LDS DMA (ASYNCcnt), double buffered
//          - 4(M)x2(N) wave tiling, LDS cross-wave argmax merge
// Stage 4: index->flow, shifted9 at scales 1/2/4, mv expansion
// Stage 5: VGG head: conv1_1 direct; conv1_2/2_1/2_2/3_1 as f16 WMMA
//          implicit GEMM (NHWC f16 activations, packed f16 weights)
// ---------------------------------------------------------------------------

typedef __attribute__((ext_vector_type(16))) _Float16 v16h;
typedef __attribute__((ext_vector_type(8)))  float    v8f;

union Frag16 { v16h v; uint4 q[2]; };

#define BATCH   2
#define FH      80
#define FW      80
#define FC      256
#define WM      78
#define PATCHES 6084        // 78*78
#define KDIM    2304        // 256*9
#define KSTEPS  72          // 2304/32
#define MTILE   128
#define NTILE   128
#define NBLKS   48          // ceil(6084/128)
#define ROWPAD  40          // 32 + 8 pad, ushort units (80B row stride, 16B aligned)

__device__ __forceinline__ unsigned short f32_to_h(float x) {
  _Float16 h = (_Float16)x;
  unsigned short u;
  __builtin_memcpy(&u, &h, 2);
  return u;
}
__device__ __forceinline__ float h_to_f32(unsigned short u) {
  _Float16 h;
  __builtin_memcpy(&h, &u, 2);
  return (float)h;
}

// ---------------- Stage 1: channel normalization -> f16 --------------------
__global__ void k_normalize(const float* __restrict__ f,
                            unsigned short* __restrict__ outh,
                            float* __restrict__ locsq) {
  const int loc = blockIdx.x;       // 0..6399
  const int b   = blockIdx.y;       // 0..1
  const int c   = threadIdx.x;      // 0..255
  float v = f[((long)b * FC + c) * (FH * FW) + loc];
  float s = v * v;
  #pragma unroll
  for (int m = 16; m; m >>= 1) s += __shfl_xor(s, m, 32);
  __shared__ float red[8];
  const int lane = threadIdx.x & 31, wv = threadIdx.x >> 5;
  if (lane == 0) red[wv] = s;
  __syncthreads();
  float tot = 0.f;
  #pragma unroll
  for (int i = 0; i < 8; ++i) tot += red[i];
  const float nrm   = sqrtf(tot);
  const float scale = 1.0f / fmaxf(nrm, 1e-12f);
  outh[((long)b * (FH * FW) + loc) * FC + c] = f32_to_h(v * scale);
  if (threadIdx.x == 0) locsq[b * (FH * FW) + loc] = tot * scale * scale;
}

// ---------------- Stage 2: 3x3 patch norms ---------------------------------
__global__ void k_patchnorm(const float* __restrict__ locsq,
                            float* __restrict__ pn) {
  int i = blockIdx.x * blockDim.x + threadIdx.x; // b*PATCHES + p
  if (i >= BATCH * PATCHES) return;
  const int b = i / PATCHES, p = i % PATCHES;
  const int py = p / WM, px = p % WM;
  float s = 0.f;
  #pragma unroll
  for (int dy = 0; dy < 3; ++dy)
    #pragma unroll
    for (int dx = 0; dx < 3; ++dx)
      s += locsq[b * (FH * FW) + (py + dy) * FW + (px + dx)];
  pn[i] = sqrtf(s);
}

// ---------------- Stage 3: fused WMMA corr GEMM + argmax -------------------
__global__ __launch_bounds__(256, 1)
void k_corr_argmax(const unsigned short* __restrict__ nA,
                   const unsigned short* __restrict__ nB,
                   const float* __restrict__ pinnorm,
                   const float* __restrict__ prefnorm,
                   int* __restrict__ maxidx,
                   float* __restrict__ maxval) {
  __shared__ __align__(16) unsigned short smA[2][MTILE * ROWPAD];
  __shared__ __align__(16) unsigned short smB[2][NTILE * ROWPAD];
  __shared__ float smRedV[2 * MTILE];
  __shared__ int   smRedC[2 * MTILE];

  const int b     = blockIdx.y;
  const int mBase = blockIdx.x * MTILE;
  const int tid   = threadIdx.x;
  const int lane  = tid & 31;
  const int wave  = tid >> 5;
  const int wm    = wave >> 1;     // 0..3 : 32-row M slice
  const int wn    = wave & 1;      // 0..1 : 64-col N half
  const int ldRow = tid & 127;     // loader row within tile
  const int ldMat = tid >> 7;      // 0 = A, 1 = B

  const unsigned short* gA = nA + (long)b * FH * FW * FC;
  const unsigned short* gB = nB + (long)b * FH * FW * FC;

  // per-thread LDS destination byte addresses (low 32 bits of flat = LDS ofs)
  unsigned ldsDst[2];
  {
    unsigned short* p0 = (ldMat ? &smB[0][0] : &smA[0][0]) + ldRow * ROWPAD;
    unsigned short* p1 = (ldMat ? &smB[1][0] : &smA[1][0]) + ldRow * ROWPAD;
    ldsDst[0] = (unsigned)(unsigned long long)p0;
    ldsDst[1] = (unsigned)(unsigned long long)p1;
  }

  float runV = -3.0e38f;   // row tid's running max (tid < 128)
  int   runC = 0;

  for (int nb = 0; nb < NBLKS; ++nb) {
    const int nBase = nb * NTILE;

    int rowp = (ldMat ? nBase : mBase) + ldRow;
    if (rowp >= PATCHES) rowp = PATCHES - 1;
    const int ry = rowp / WM, rx = rowp - ry * WM;
    const unsigned short* gsrc =
        (ldMat ? gB : gA) + ((long)(ry * FW + rx)) * FC;

    auto issue = [&](int buf, int ks) {
      const int dy    = ks / 24;               // which 3x1 segment
      const int inner = (ks - dy * 24) * 32;   // element offset in segment
      const unsigned long long ga =
          (unsigned long long)(gsrc + (long)dy * (FW * FC) + inner);
      const unsigned ld = ldsDst[buf];
      // IOFFSET applies to both LDS and global address (ISA 08 s4.4)
      asm volatile("global_load_async_to_lds_b128 %0, %1, off"
                   :: "v"(ld), "v"(ga) : "memory");
      asm volatile("global_load_async_to_lds_b128 %0, %1, off offset:16"
                   :: "v"(ld), "v"(ga) : "memory");
      asm volatile("global_load_async_to_lds_b128 %0, %1, off offset:32"
                   :: "v"(ld), "v"(ga) : "memory");
      asm volatile("global_load_async_to_lds_b128 %0, %1, off offset:48"
                   :: "v"(ld), "v"(ga) : "memory");
    };

    v8f acc[2][4];
    #pragma unroll
    for (int mt = 0; mt < 2; ++mt)
      #pragma unroll
      for (int nt = 0; nt < 4; ++nt)
        #pragma unroll
        for (int j = 0; j < 8; ++j) acc[mt][nt][j] = 0.0f;

    issue(0, 0);   // prologue fill of buffer 0

    for (int ks = 0; ks < KSTEPS; ++ks) {
      const int cur = ks & 1;
      asm volatile("s_wait_asynccnt 0x0" ::: "memory");
      __syncthreads();
      if (ks + 1 < KSTEPS) issue(cur ^ 1, ks + 1);   // overlap next fill

      Frag16 a[2];
      #pragma unroll
      for (int mt = 0; mt < 2; ++mt) {
        const int row   = wm * 32 + mt * 16 + (lane & 15);
        const int halfK = (lane & 16) ? 16 : 0;
        const unsigned short* sp = &smA[cur][0] + row * ROWPAD + halfK;
        a[mt].q[0] = *(const uint4*)(sp);
        a[mt].q[1] = *(const uint4*)(sp + 8);
      }
      #pragma unroll
      for (int nt = 0; nt < 4; ++nt) {
        Frag16 bf;
        const int col = wn * 64 + nt * 16 + (lane & 15);
        const int kb  = (lane & 16) ? 16 : 0;
        const unsigned short* sp = &smB[cur][0] + col * ROWPAD + kb;
        bf.q[0] = *(const uint4*)(sp);
        bf.q[1] = *(const uint4*)(sp + 8);
        #pragma unroll
        for (int mt = 0; mt < 2; ++mt)
          acc[mt][nt] = __builtin_amdgcn_wmma_f32_16x16x32_f16(
              false, a[mt].v, false, bf.v, (short)0, acc[mt][nt], false, false);
      }
    }

    // epilogue: scale by 1/(||pref||+1e-5), per-row max/argmax
    float bv[2][8]; int bc[2][8];
    #pragma unroll
    for (int mt = 0; mt < 2; ++mt)
      #pragma unroll
      for (int v = 0; v < 8; ++v) { bv[mt][v] = -3.0e38f; bc[mt][v] = 0; }

    #pragma unroll
    for (int nt = 0; nt < 4; ++nt) {
      const int q   = nBase + wn * 64 + nt * 16 + (lane & 15);
      const bool ok = (q < PATCHES);
      const float inv = ok ? 1.0f / (prefnorm[b * PATCHES + q] + 1e-5f) : 0.f;
      #pragma unroll
      for (int mt = 0; mt < 2; ++mt)
        #pragma unroll
        for (int v = 0; v < 8; ++v) {
          const float val = ok ? acc[mt][nt][v] * inv : -3.0e38f;
          if (val > bv[mt][v]) { bv[mt][v] = val; bc[mt][v] = q; }
        }
    }
    #pragma unroll
    for (int m = 1; m < 16; m <<= 1) {
      #pragma unroll
      for (int mt = 0; mt < 2; ++mt)
        #pragma unroll
        for (int v = 0; v < 8; ++v) {
          const float ov = __shfl_xor(bv[mt][v], m, 32);
          const int   oc = __shfl_xor(bc[mt][v], m, 32);
          if (ov > bv[mt][v] || (ov == bv[mt][v] && oc < bc[mt][v])) {
            bv[mt][v] = ov; bc[mt][v] = oc;
          }
        }
    }
    if ((lane & 15) == 0) {
      const int hi = (lane >> 4) & 1;
      #pragma unroll
      for (int mt = 0; mt < 2; ++mt)
        #pragma unroll
        for (int v = 0; v < 8; ++v) {
          const int row = wm * 32 + mt * 16 + v + 8 * hi;
          smRedV[wn * MTILE + row] = bv[mt][v];
          smRedC[wn * MTILE + row] = bc[mt][v];
        }
    }
    __syncthreads();
    if (tid < MTILE) {
      float v0 = smRedV[tid];          int c0 = smRedC[tid];
      const float v1 = smRedV[MTILE + tid];
      const int   c1 = smRedC[MTILE + tid];
      if (v1 > v0 || (v1 == v0 && c1 < c0)) { v0 = v1; c0 = c1; }
      if (v0 > runV || (v0 == runV && c0 < runC)) { runV = v0; runC = c0; }
    }
    __syncthreads();
  }

  if (tid < MTILE) {
    const int p = mBase + tid;
    if (p < PATCHES) {
      maxidx[b * PATCHES + p] = runC;
      maxval[b * PATCHES + p] = runV / (pinnorm[b * PATCHES + p] + 1e-5f);
    }
  }
}

// ---------------- Stage 4: flows / shifts / mv -----------------------------
__global__ void k_shifted(const int* __restrict__ maxidx,
                          float* __restrict__ out, int r) {
  const int  Hs    = FH * r;
  const long total = (long)BATCH * 9 * Hs * Hs * 2;
  const long stride = (long)gridDim.x * blockDim.x;
  for (long i = (long)blockIdx.x * blockDim.x + threadIdx.x; i < total; i += stride) {
    const int c = (int)(i & 1);
    long t = i >> 1;
    const int x = (int)(t % Hs); t /= Hs;
    const int y = (int)(t % Hs); t /= Hs;
    const int s = (int)(t % 9);  t /= 9;
    const int b = (int)t;
    const int si = s / 3, sj = s % 3;
    float val = 0.0f;
    const int yy = y - si * r, xx = x - sj * r;
    if (yy >= 0 && xx >= 0) {
      const int oy = yy / r, ox = xx / r;
      if (oy < WM && ox < WM) {
        const int idx = maxidx[b * PATCHES + oy * WM + ox];
        val = (c == 0 ? (float)(idx % WM - ox) : (float)(idx / WM - oy)) * (float)r;
      }
    }
    out[i] = val;
  }
}

__global__ void k_mv(const float* __restrict__ maxval, float* __restrict__ out) {
  const long total = (long)BATCH * 320 * 320;
  const long stride = (long)gridDim.x * blockDim.x;
  for (long i = (long)blockIdx.x * blockDim.x + threadIdx.x; i < total; i += stride) {
    const int X = (int)(i % 320);
    long t = i / 320;
    const int Y = (int)(t % 320);
    const int b = (int)(t / 320);
    const int y4 = Y >> 2, x4 = X >> 2;
    float v = 0.0f;
    if (y4 >= 1 && y4 <= WM && x4 >= 1 && x4 <= WM)
      v = maxval[b * PATCHES + (y4 - 1) * WM + (x4 - 1)];
    out[i] = v;
  }
}

// ---------------- Stage 5: VGG head ----------------------------------------
__global__ void k_prep(const float* __restrict__ img, float* __restrict__ out) {
  const long total = (long)BATCH * 3 * 320 * 320;
  const long stride = (long)gridDim.x * blockDim.x;
  const float mean[3] = {0.485f, 0.456f, 0.406f};
  const float stdv[3] = {0.229f, 0.224f, 0.225f};
  for (long i = (long)blockIdx.x * blockDim.x + threadIdx.x; i < total; i += stride) {
    const int c = (int)((i / (320 * 320)) % 3);
    out[i] = (img[i] - mean[c]) / stdv[c];
  }
}

// direct conv (used only for conv1_1, Cin=3); also emits f16 NHWC copy
__global__ void k_conv3x3_relu(const float* __restrict__ in,
                               const float* __restrict__ w,
                               const float* __restrict__ bias,
                               float* __restrict__ out,
                               unsigned short* __restrict__ outh,
                               int Cin, int Cout, int H, int W) {
  const long total = (long)BATCH * Cout * H * W;
  const long stride = (long)gridDim.x * blockDim.x;
  for (long i = (long)blockIdx.x * blockDim.x + threadIdx.x; i < total; i += stride) {
    const int x = (int)(i % W);
    long t = i / W;
    const int y  = (int)(t % H); t /= H;
    const int oc = (int)(t % Cout);
    const int b  = (int)(t / Cout);
    float s = bias[oc];
    const float* wp = w + (long)oc * Cin * 9;
    const float* ip = in + (long)b * Cin * H * W;
    for (int ic = 0; ic < Cin; ++ic) {
      const float* ir = ip + (long)ic * H * W;
      const float* wr = wp + ic * 9;
      #pragma unroll
      for (int dy = 0; dy < 3; ++dy) {
        const int iy = y + dy - 1;
        if (iy < 0 || iy >= H) continue;
        #pragma unroll
        for (int dx = 0; dx < 3; ++dx) {
          const int ix = x + dx - 1;
          if (ix < 0 || ix >= W) continue;
          s = fmaf(ir[(long)iy * W + ix], wr[dy * 3 + dx], s);
        }
      }
    }
    const float r = fmaxf(s, 0.0f);
    out[i] = r;
    if (outh)
      outh[((long)b * H * W + (long)y * W + x) * Cout + oc] = f32_to_h(r);
  }
}

// pack OIHW fp32 weights -> f16 [oc][k], k = (dy*3+dx)*Cin + ic
__global__ void k_packw(const float* __restrict__ w,
                        unsigned short* __restrict__ wp, int Cin, int Cout) {
  const int K = Cin * 9;
  const int i = blockIdx.x * blockDim.x + threadIdx.x; // oc*K + k
  if (i >= Cout * K) return;
  const int oc = i / K, k = i % K;
  const int seg = k / Cin, ic = k % Cin;
  const int dy = seg / 3, dx = seg % 3;
  wp[i] = f32_to_h(w[((long)(oc * Cin + ic) * 3 + dy) * 3 + dx]);
}

// WMMA implicit-GEMM 3x3 SAME conv + bias + relu.
// A = packed weights [oc][K], B = im2col of f16 NHWC activations.
// Block: 64(oc) x 128(spatial) tile, 8 waves as 2(M)x4(N); K in steps of 32
// (each step = one contiguous 64B span per row: Cin is a multiple of 32).
__global__ __launch_bounds__(256, 1)
void k_convw(const unsigned short* __restrict__ inh,   // (b, y, x, ic) f16
             const unsigned short* __restrict__ wp,    // (oc, k) f16
             const float* __restrict__ bias,
             float* __restrict__ out32,                // NCHW fp32 (or null)
             unsigned short* __restrict__ outh,        // NHWC f16 (or null)
             int Cin, int Cout, int H, int W) {
  __shared__ __align__(16) unsigned short sA[64 * ROWPAD];
  __shared__ __align__(16) unsigned short sB[128 * ROWPAD];

  const int b     = blockIdx.z;
  const int ocg   = blockIdx.y;            // 64-wide Cout slice
  const int nBase = blockIdx.x * 128;      // spatial base (H*W % 128 == 0)
  const int tid = threadIdx.x, lane = tid & 31, wave = tid >> 5;
  const int wm = wave >> 2;                // 0..1 : 32-row M slice
  const int wn = wave & 3;                 // 0..3 : 32-col N slice
  const int K         = Cin * 9;
  const int ksteps    = K >> 5;
  const int segBlocks = Cin >> 5;          // 32-wide k-blocks per 3x3 segment

  const unsigned short* inB = inh + (long)b * H * W * Cin;

  const bool isA = (tid < 64);
  const bool isB = (tid >= 64 && tid < 192);
  const int aRow = tid;
  const int bRow = tid - 64;
  const int nPos = nBase + bRow;
  const int by = nPos / W, bx = nPos % W;

  v8f acc[2][2];
  #pragma unroll
  for (int mt = 0; mt < 2; ++mt)
    #pragma unroll
    for (int nt = 0; nt < 2; ++nt)
      #pragma unroll
      for (int j = 0; j < 8; ++j) acc[mt][nt][j] = 0.0f;

  for (int kb = 0; kb < ksteps; ++kb) {
    const int seg     = kb / segBlocks;
    const int innerIC = (kb - seg * segBlocks) * 32;
    const int dy = seg / 3, dx = seg % 3;
    __syncthreads();
    if (isA) {
      const uint4* src = (const uint4*)(wp + (long)(ocg * 64 + aRow) * K + kb * 32);
      uint4 d0 = src[0], d1 = src[1], d2 = src[2], d3 = src[3];
      unsigned short* dst = sA + aRow * ROWPAD;
      *(uint4*)(dst + 0)  = d0;
      *(uint4*)(dst + 8)  = d1;
      *(uint4*)(dst + 16) = d2;
      *(uint4*)(dst + 24) = d3;
    } else if (isB) {
      const int iy = by + dy - 1, ix = bx + dx - 1;
      const bool valid = (iy >= 0 && iy < H && ix >= 0 && ix < W);
      uint4 z; z.x = z.y = z.z = z.w = 0u;
      uint4 d0 = z, d1 = z, d2 = z, d3 = z;
      if (valid) {
        const uint4* src =
            (const uint4*)(inB + ((long)iy * W + ix) * Cin + innerIC);
        d0 = src[0]; d1 = src[1]; d2 = src[2]; d3 = src[3];
      }
      unsigned short* dst = sB + bRow * ROWPAD;
      *(uint4*)(dst + 0)  = d0;
      *(uint4*)(dst + 8)  = d1;
      *(uint4*)(dst + 16) = d2;
      *(uint4*)(dst + 24) = d3;
    }
    __syncthreads();

    Frag16 a[2];
    #pragma unroll
    for (int mt = 0; mt < 2; ++mt) {
      const int row   = wm * 32 + mt * 16 + (lane & 15);
      const int halfK = (lane & 16) ? 16 : 0;
      const unsigned short* sp = sA + row * ROWPAD + halfK;
      a[mt].q[0] = *(const uint4*)(sp);
      a[mt].q[1] = *(const uint4*)(sp + 8);
    }
    #pragma unroll
    for (int nt = 0; nt < 2; ++nt) {
      Frag16 bf;
      const int col = wn * 32 + nt * 16 + (lane & 15);
      const int kb2 = (lane & 16) ? 16 : 0;
      const unsigned short* sp = sB + col * ROWPAD + kb2;
      bf.q[0] = *(const uint4*)(sp);
      bf.q[1] = *(const uint4*)(sp + 8);
      #pragma unroll
      for (int mt = 0; mt < 2; ++mt)
        acc[mt][nt] = __builtin_amdgcn_wmma_f32_16x16x32_f16(
            false, a[mt].v, false, bf.v, (short)0, acc[mt][nt], false, false);
    }
  }

  // epilogue: bias + relu, dual write (fp32 NCHW and/or f16 NHWC)
  const int hi = (lane >> 4) & 1;
  #pragma unroll
  for (int mt = 0; mt < 2; ++mt)
    #pragma unroll
    for (int nt = 0; nt < 2; ++nt) {
      const int col = nBase + wn * 32 + nt * 16 + (lane & 15);
      #pragma unroll
      for (int v = 0; v < 8; ++v) {
        const int row = wm * 32 + mt * 16 + v + 8 * hi;
        const int oc  = ocg * 64 + row;
        const float val = fmaxf(acc[mt][nt][v] + bias[oc], 0.0f);
        if (out32) out32[((long)(b * Cout + oc)) * (H * W) + col] = val;
        if (outh)  outh[((long)b * H * W + col) * Cout + oc] = f32_to_h(val);
      }
    }
}

// 2x2 max-pool on f16 NHWC
__global__ void k_pool2_h(const unsigned short* __restrict__ in,
                          unsigned short* __restrict__ out,
                          int C, int H, int W) {
  const int Ho = H / 2, Wo = W / 2;
  const long total = (long)BATCH * Ho * Wo * C;
  const long stride = (long)gridDim.x * blockDim.x;
  for (long i = (long)blockIdx.x * blockDim.x + threadIdx.x; i < total; i += stride) {
    const int c = (int)(i % C);
    long t = i / C;
    const int x = (int)(t % Wo); t /= Wo;
    const int y = (int)(t % Ho);
    const int b = (int)(t / Ho);
    const long base = (long)b * H * W * C;
    const float v00 = h_to_f32(in[base + ((long)(2 * y) * W + 2 * x) * C + c]);
    const float v01 = h_to_f32(in[base + ((long)(2 * y) * W + 2 * x + 1) * C + c]);
    const float v10 = h_to_f32(in[base + ((long)(2 * y + 1) * W + 2 * x) * C + c]);
    const float v11 = h_to_f32(in[base + ((long)(2 * y + 1) * W + 2 * x + 1) * C + c]);
    out[i] = f32_to_h(fmaxf(fmaxf(v00, v01), fmaxf(v10, v11)));
  }
}

// ---------------------------------------------------------------------------
static inline int nblk(long n) {
  long b = (n + 255) / 256;
  if (b > 262144) b = 262144;
  return (int)b;
}

extern "C" void kernel_launch(void* const* d_in, const int* in_sizes, int n_in,
                              void* d_out, int out_size, void* d_ws, size_t ws_size,
                              hipStream_t stream) {
  const float* f1  = (const float*)d_in[0];
  const float* f2  = (const float*)d_in[1];
  const float* img = (const float*)d_in[8];
  const float* w11 = (const float*)d_in[9];
  const float* b11 = (const float*)d_in[10];
  const float* w12 = (const float*)d_in[11];
  const float* b12 = (const float*)d_in[12];
  const float* w21 = (const float*)d_in[13];
  const float* b21 = (const float*)d_in[14];
  const float* w22 = (const float*)d_in[15];
  const float* b22 = (const float*)d_in[16];
  const float* w31 = (const float*)d_in[17];
  const float* b31 = (const float*)d_in[18];

  char* ws = (char*)d_ws;
  unsigned short* normA = (unsigned short*)(ws + 0);            // 6.55 MB
  unsigned short* normB = (unsigned short*)(ws + (4LL  << 20)); // 6.55 MB
  float* locsqA  = (float*)(ws + (8LL  << 20));
  float* locsqB  = (float*)(ws + (9LL  << 20));
  float* pinnorm = (float*)(ws + (10LL << 20));
  float* prefnorm= (float*)(ws + (11LL << 20));
  int*   maxidx  = (int*)  (ws + (12LL << 20));
  float* maxval  = (float*)(ws + (13LL << 20));
  float* vx = (float*)(ws + (16LL << 20));                      //  2.46 MB
  unsigned short* r11h = (unsigned short*)(ws + (20LL  << 20)); // 26.3 MB
  unsigned short* t1h  = (unsigned short*)(ws + (48LL  << 20)); // 26.3 MB
  unsigned short* t2h  = (unsigned short*)(ws + (76LL  << 20)); //  6.6 MB
  unsigned short* r21h = (unsigned short*)(ws + (84LL  << 20)); // 13.2 MB
  unsigned short* t3h  = (unsigned short*)(ws + (98LL  << 20)); // 13.2 MB
  unsigned short* t4h  = (unsigned short*)(ws + (112LL << 20)); //  3.3 MB
  unsigned short* w12p = (unsigned short*)(ws + (116LL << 20)); //   73 KB
  unsigned short* w21p = (unsigned short*)(ws + (117LL << 20)); //  147 KB
  unsigned short* w22p = (unsigned short*)(ws + (118LL << 20)); //  294 KB
  unsigned short* w31p = (unsigned short*)(ws + (119LL << 20)); //  590 KB

  float* out   = (float*)d_out;   // tree-flatten order: sh1, sh2, sh3, r11, r21, r31, mv
  float* o_sh1 = out;             // (2,9,320,320,2) = 3,686,400
  float* o_sh2 = out + 3686400;   // (2,9,160,160,2) =   921,600
  float* o_sh3 = out + 4608000;   // (2,9, 80, 80,2) =   230,400
  float* o_r11 = out + 4838400;   // (2, 64,320,320) = 13,107,200
  float* o_r21 = out + 17945600;  // (2,128,160,160) =  6,553,600
  float* o_r31 = out + 24499200;  // (2,256, 80, 80) =  3,276,800
  float* o_mv  = out + 27776000;  // (2,  1,320,320) =    204,800

  // Stage 1/2: normalization + patch norms
  k_normalize<<<dim3(FH * FW, BATCH), 256, 0, stream>>>(f1, normA, locsqA);
  k_normalize<<<dim3(FH * FW, BATCH), 256, 0, stream>>>(f2, normB, locsqB);
  k_patchnorm<<<(BATCH * PATCHES + 255) / 256, 256, 0, stream>>>(locsqA, pinnorm);
  k_patchnorm<<<(BATCH * PATCHES + 255) / 256, 256, 0, stream>>>(locsqB, prefnorm);

  // Stage 3: fused WMMA correlation GEMM + streaming argmax
  k_corr_argmax<<<dim3(NBLKS, BATCH), 256, 0, stream>>>(
      normA, normB, pinnorm, prefnorm, maxidx, maxval);

  // Stage 4: flows / shifts / mv
  k_shifted<<<nblk((long)BATCH * 9 * 320 * 320 * 2), 256, 0, stream>>>(maxidx, o_sh1, 4);
  k_shifted<<<nblk((long)BATCH * 9 * 160 * 160 * 2), 256, 0, stream>>>(maxidx, o_sh2, 2);
  k_shifted<<<nblk((long)BATCH * 9 *  80 *  80 * 2), 256, 0, stream>>>(maxidx, o_sh3, 1);
  k_mv<<<nblk((long)BATCH * 320 * 320), 256, 0, stream>>>(maxval, o_mv);

  // Stage 5: VGG head
  k_packw<<<nblk((long)64  * 576),  256, 0, stream>>>(w12, w12p, 64, 64);
  k_packw<<<nblk((long)128 * 576),  256, 0, stream>>>(w21, w21p, 64, 128);
  k_packw<<<nblk((long)128 * 1152), 256, 0, stream>>>(w22, w22p, 128, 128);
  k_packw<<<nblk((long)256 * 1152), 256, 0, stream>>>(w31, w31p, 128, 256);
  k_prep<<<nblk((long)BATCH * 3 * 320 * 320), 256, 0, stream>>>(img, vx);

  // conv1_1: direct (K=27, negligible FLOPs); emits fp32 NCHW + f16 NHWC
  k_conv3x3_relu<<<nblk((long)BATCH * 64 * 320 * 320), 256, 0, stream>>>(
      vx, w11, b11, o_r11, r11h, 3, 64, 320, 320);
  // conv1_2 (WMMA): 64->64 @320^2, f16 NHWC only (feeds pool)
  k_convw<<<dim3(800, 1, BATCH), 256, 0, stream>>>(
      r11h, w12p, b12, nullptr, t1h, 64, 64, 320, 320);
  k_pool2_h<<<nblk((long)BATCH * 64 * 160 * 160), 256, 0, stream>>>(
      t1h, t2h, 64, 320, 320);
  // conv2_1 (WMMA): 64->128 @160^2, fp32 NCHW output + f16 NHWC
  k_convw<<<dim3(200, 2, BATCH), 256, 0, stream>>>(
      t2h, w21p, b21, o_r21, r21h, 64, 128, 160, 160);
  // conv2_2 (WMMA): 128->128 @160^2, f16 NHWC only
  k_convw<<<dim3(200, 2, BATCH), 256, 0, stream>>>(
      r21h, w22p, b22, nullptr, t3h, 128, 128, 160, 160);
  k_pool2_h<<<nblk((long)BATCH * 128 * 80 * 80), 256, 0, stream>>>(
      t3h, t4h, 128, 160, 160);
  // conv3_1 (WMMA): 128->256 @80^2, fp32 NCHW output only
  k_convw<<<dim3(50, 4, BATCH), 256, 0, stream>>>(
      t4h, w31p, b31, o_r31, nullptr, 128, 256, 80, 80);
}
